// GRNGOB_53609781788995
// MI455X (gfx1250) — compile-verified, matrix operand order checked by reading, not run on hardware
//
#include <hip/hip_runtime.h>
#include <cstddef>
#include <cstdint>

// ---------------- problem constants ----------------
#define BB   8
#define SS   64
#define WW   128
#define HH   128     // SE = EE = SH = EH = G = 128
#define LL   32
#define NG   160     // gate output dim (EH+L or SH+L)
#define RZS  672     // s_gru rz dim
#define XS   544     // s_gru x dim
#define RZE  544     // e_gru rz dim
#define XE   416     // e_gru x dim

#define KT   32      // GEMM K tile (all K are multiples of 32)
#define USE_ASYNC_LDS 1

typedef __attribute__((ext_vector_type(16))) __bf16 bf16x16;
typedef __attribute__((ext_vector_type(8)))  __bf16 bf16x8;
typedef __attribute__((ext_vector_type(8)))  float  f32x8;

__device__ __forceinline__ float sigf(float x) { return 1.0f / (1.0f + __expf(-x)); }

// ---------------- WMMA GEMM (double-buffered async LDS A panel) ----------------
// C[M,N](f32,row,ldc) = act( A[M,K](bf16,row,lda) x Wt[N,K](bf16,row,ldw)^T + bias[N] )
// Block = 4 waves. Block owns a 64-row M panel; wave w owns N-tile (tnBase+w).
// Pipeline: async-copy panel(k+1) global->LDS while computing WMMAs on panel(k);
// one s_wait_asynccnt + one barrier per K-step. M % 64 == 0, N % 16 == 0, K % KT == 0.
__global__ void k_wmma_gemm(const __bf16* __restrict__ A, int lda,
                            const __bf16* __restrict__ Wt, int ldw,
                            const float* __restrict__ bias,
                            float* __restrict__ C, int ldc,
                            int M, int N, int K, int act) {
  __shared__ __attribute__((aligned(16))) __bf16 panel[2][64 * KT];   // 2 x 4 KB
  const int lane = threadIdx.x & 31;
  const int wave = threadIdx.x >> 5;
  const int tilesN = N >> 4;
  const int tnGroups = (tilesN + 3) >> 2;
  const int m0 = (blockIdx.x / tnGroups) * 64;
  const int tn = (blockIdx.x % tnGroups) * 4 + wave;
  const bool valid = (tn < tilesN) && (m0 < M);
  const int l = lane & 15, hi = lane >> 4;
  // B fragment source: row (tn*16+l) of Wt, K chunks [hi*8, +8) and [16+hi*8, +8)
  const __bf16* pb = Wt + (size_t)((valid ? tn : 0) * 16 + l) * ldw + hi * 8;
  const unsigned ldsBase = (unsigned)(uintptr_t)(void*)&panel[0][0];  // LDS offset = low 32 bits

  // stage one 64xKT panel (256 x 16B chunks, 2 per thread) into panel[buf]
  auto stage = [&](int k0, int buf) {
#pragma unroll
    for (int c = threadIdx.x; c < (64 * KT * 2) / 16; c += 128) {
      const int row = c >> 2;
      const int col = (c & 3) * 8;
      const __bf16* src = A + (size_t)(m0 + row) * lda + k0 + col;
#if USE_ASYNC_LDS
      unsigned dst = ldsBase + (unsigned)buf * (64 * KT * 2) + c * 16;
      asm volatile("global_load_async_to_lds_b128 %0, %1, off"
                   :: "v"(dst), "v"(src) : "memory");
#else
      *(bf16x8*)&panel[buf][row * KT + col] = *(const bf16x8*)src;
#endif
    }
  };

  f32x8 acc[4] = {{}, {}, {}, {}};
  const int nsteps = K / KT;

  stage(0, 0);
#if USE_ASYNC_LDS
  asm volatile("s_wait_asynccnt 0" ::: "memory");
#endif
  __syncthreads();

  for (int i = 0; i < nsteps; ++i) {
    const int buf = i & 1;
    if (i + 1 < nsteps) stage((i + 1) * KT, buf ^ 1);   // prefetch next panel (async)
    if (valid) {
      const int k0 = i * KT;
      union { bf16x16 v; bf16x8 h[2]; } b;
      b.h[0] = *(const bf16x8*)(pb + k0);
      b.h[1] = *(const bf16x8*)(pb + k0 + 16);
      __builtin_prefetch(pb + k0 + KT, 0, 3);
#pragma unroll
      for (int mt = 0; mt < 4; ++mt) {
        union { bf16x16 v; bf16x8 h[2]; } a;
        const __bf16* pl = &panel[buf][(mt * 16 + l) * KT + hi * 8];
        a.h[0] = *(const bf16x8*)(pl);
        a.h[1] = *(const bf16x8*)(pl + 16);
        acc[mt] = __builtin_amdgcn_wmma_f32_16x16x32_bf16(false, a.v, false, b.v,
                                                          (short)0, acc[mt], false, false);
      }
    }
#if USE_ASYNC_LDS
    asm volatile("s_wait_asynccnt 0" ::: "memory");   // next panel landed (per-wave)
#endif
    __syncthreads();                                  // ... and visible to all waves
  }

  if (valid) {
    const int n = tn * 16 + l;
    const float bv = bias ? bias[n] : 0.0f;
#pragma unroll
    for (int mt = 0; mt < 4; ++mt) {
#pragma unroll
      for (int r = 0; r < 8; ++r) {
        int m = m0 + mt * 16 + hi * 8 + r;   // C/D layout: vgpr r, lanes16-31 -> M=r+8
        float v = acc[mt][r] + bv;
        if (act == 1) v = sigf(v);
        C[(size_t)m * ldc + n] = v;
      }
    }
  }
}

// ---------------- small utility kernels ----------------
__global__ void k_zero_f32(float* p, int n) {
  int i = blockIdx.x * blockDim.x + threadIdx.x;
  if (i < n) p[i] = 0.0f;
}
__global__ void k_zero_bf16(__bf16* p, int n) {
  int i = blockIdx.x * blockDim.x + threadIdx.x;
  if (i < n) p[i] = (__bf16)0.0f;
}
// W: K x N (f32, row) -> Wt: N x K (bf16, row)
__global__ void k_transpose_bf16(const float* __restrict__ W, __bf16* __restrict__ Wt,
                                 int K, int N) {
  int i = blockIdx.x * blockDim.x + threadIdx.x;
  if (i >= K * N) return;
  int k = i / N, n = i % N;
  Wt[(size_t)n * K + k] = (__bf16)W[i];
}
// T[m][j] = edgeemb[m] @ gateW[256:288, j] + gateb[j]   (4 x 160)
__global__ void k_label_table(const float* __restrict__ edgeemb,
                              const float* __restrict__ gateW,
                              const float* __restrict__ gateb,
                              float* __restrict__ T) {
  int i = blockIdx.x * blockDim.x + threadIdx.x;
  if (i >= 4 * NG) return;
  int m = i / NG, j = i % NG;
  float acc = gateb[j];
  for (int t = 0; t < LL; ++t)
    acc += edgeemb[m * LL + t] * gateW[(size_t)(2 * HH + t) * NG + j];
  T[i] = acc;
}
// rowbits[(b,s)][t] : bit i set if matrix[b,s,t*32+i] != 0
__global__ void k_rowbits(const int* __restrict__ matrix, unsigned* __restrict__ rb) {
  int i = blockIdx.x * blockDim.x + threadIdx.x;
  if (i >= BB * SS * 4) return;
  int bs = i >> 2, t = i & 3;
  const int* mr = matrix + (size_t)bs * WW + t * 32;
  unsigned bits = 0;
#pragma unroll
  for (int k = 0; k < 32; ++k) bits |= (mr[k] != 0) ? (1u << k) : 0u;
  rb[i] = bits;
}
// s2s adjacency bits: 64 bits per (b,i): share a word, j!=i, smask[b,j]
__global__ void k_s2sbits(const unsigned* __restrict__ rb, const int* __restrict__ smask,
                          unsigned* __restrict__ s2s) {
  int i = blockIdx.x * blockDim.x + threadIdx.x;
  if (i >= BB * SS * 2) return;
  int bi = i >> 1, t = i & 1;
  int b = bi / SS, row = bi % SS;
  unsigned r0 = rb[bi * 4 + 0], r1 = rb[bi * 4 + 1], r2 = rb[bi * 4 + 2], r3 = rb[bi * 4 + 3];
  unsigned bits = 0;
  for (int jj = 0; jj < 32; ++jj) {
    int j = t * 32 + jj;
    if (j == row) continue;
    if (smask[b * SS + j] == 0) continue;
    const unsigned* q = rb + ((size_t)b * SS + j) * 4;
    if ((r0 & q[0]) | (r1 & q[1]) | (r2 & q[2]) | (r3 & q[3])) bits |= 1u << jj;
  }
  s2s[i] = bits;
}
// s_neigh_s_h[b,i,:] = sum over adjacent j of s_h[b,j,:]
__global__ void k_neighbor_sum(const unsigned* __restrict__ s2s,
                               const float* __restrict__ sH, float* __restrict__ out) {
  int bi = blockIdx.x, col = threadIdx.x;
  int b = bi / SS;
  unsigned m0 = s2s[bi * 2], m1 = s2s[bi * 2 + 1];
  const float* hb = sH + (size_t)b * SS * HH;
  float acc = 0.0f;
  for (int j = 0; j < SS; ++j) {
    unsigned bit = (j < 32) ? ((m0 >> j) & 1u) : ((m1 >> (j - 32)) & 1u);
    if (bit) acc += hb[(size_t)j * HH + col];
  }
  out[(size_t)bi * HH + col] = acc;
}
// gate1 gather: s_neigh_e_h[b,s,j] = sum_w mask * sig(P1s+P1e+T1[m]) * feat
__global__ void k_gather1(const int* __restrict__ matrix,
                          const float* __restrict__ P1s, const float* __restrict__ P1e,
                          const float* __restrict__ T1, const float* __restrict__ eH,
                          const float* __restrict__ edgeemb, float* __restrict__ out) {
  int bs = blockIdx.x, j = threadIdx.x;      // blockDim = 160 = 5 waves
  int b = bs / SS;
  const int* mrow = matrix + (size_t)bs * WW;
  float p1s = P1s[(size_t)bs * NG + j];
  float acc = 0.0f;
  for (int w = 0; w < WW; ++w) {
    int m = mrow[w];
    if (m != 0) {
      float g = sigf(p1s + P1e[((size_t)b * WW + w) * NG + j] + T1[m * NG + j]);
      float feat = (j < HH) ? eH[((size_t)b * WW + w) * HH + j]
                            : edgeemb[m * LL + (j - HH)];
      acc += feat * g;
    }
  }
  out[(size_t)bs * NG + j] = acc;
}
// gate2 gather: e_neigh_s_h[b,w,j] = sum_s mask * sig(P2e+P2s+T2[m]) * feat
__global__ void k_gather2(const int* __restrict__ matrix,
                          const float* __restrict__ P2s, const float* __restrict__ P2e,
                          const float* __restrict__ T2, const float* __restrict__ sH,
                          const float* __restrict__ edgeemb, float* __restrict__ out) {
  int bw = blockIdx.x, j = threadIdx.x;
  int b = bw / WW, w = bw % WW;
  float p2e = P2e[(size_t)bw * NG + j];
  float acc = 0.0f;
  for (int s = 0; s < SS; ++s) {
    int m = matrix[((size_t)b * SS + s) * WW + w];
    if (m != 0) {
      float g = sigf(p2e + P2s[((size_t)b * SS + s) * NG + j] + T2[m * NG + j]);
      float feat = (j < HH) ? sH[((size_t)b * SS + s) * HH + j]
                            : edgeemb[m * LL + (j - HH)];
      acc += feat * g;
    }
  }
  out[(size_t)bw * NG + j] = acc;
}
// rz_s = [sent | s_neigh_s_h | s_neigh_e_h | g_h | s_h]  (bf16, 512 x 672)
__global__ void k_concat_rzs(const float* sent, const float* sns, const float* sne,
                             const float* gH, const float* sH, __bf16* rz) {
  int i = blockIdx.x * blockDim.x + threadIdx.x;
  if (i >= BB * SS * RZS) return;
  int row = i / RZS, c = i % RZS, b = row / SS;
  float v;
  if      (c < 128) v = sent[(size_t)row * HH + c];
  else if (c < 256) v = sns[(size_t)row * HH + (c - 128)];
  else if (c < 416) v = sne[(size_t)row * NG + (c - 256)];
  else if (c < 544) v = gH[(size_t)b * HH + (c - 416)];
  else              v = sH[(size_t)row * HH + (c - 544)];
  rz[i] = (__bf16)v;
}
// rz_e = [word | e_neigh_s_h | g_h | e_h]  (bf16, 1024 x 544)
__global__ void k_concat_rze(const float* word, const float* ens,
                             const float* gH, const float* eH, __bf16* rz) {
  int i = blockIdx.x * blockDim.x + threadIdx.x;
  if (i >= BB * WW * RZE) return;
  int row = i / RZE, c = i % RZE, b = row / WW;
  float v;
  if      (c < 128) v = word[(size_t)row * HH + c];
  else if (c < 288) v = ens[(size_t)row * NG + (c - 128)];
  else if (c < 416) v = gH[(size_t)b * HH + (c - 288)];
  else              v = eH[(size_t)row * HH + (c - 416)];
  rz[i] = (__bf16)v;
}
// GRU pointwise: new = mask * (z*h + (1-z)*tanh(C + r*U0)); also emit bf16 mirror
__global__ void k_gru_combine(const float* R, const float* Z, const float* C,
                              const float* U0, const float* H, const int* mask,
                              float* nH, __bf16* nHb, int total) {
  int i = blockIdx.x * blockDim.x + threadIdx.x;
  if (i >= total) return;
  int row = i / HH;
  float r = R[i], z = Z[i];
  float u = tanhf(C[i] + r * U0[i]);
  float v = z * H[i] + (1.0f - z) * u;
  if (mask[row] == 0) v = 0.0f;
  nH[i] = v;
  nHb[i] = (__bf16)v;
}
// column means over masked rows
__global__ void k_mean(const float* H, const int* mask, float* out, int rows, float addone) {
  int b = blockIdx.x, col = threadIdx.x;
  float s = 0.0f;
  for (int r = 0; r < rows; ++r) s += H[((size_t)b * rows + r) * HH + col];
  int cnt = 0;
  for (int r = 0; r < rows; ++r) cnt += (mask[b * rows + r] != 0);
  out[(size_t)b * HH + col] = s / ((float)cnt + addone);
}
// tiny global GRU (8 rows) — LDS-staged dot products, g_h updated in place
__global__ void k_ggru(const float* shM, const float* ehM,
                       const float* Wr, const float* Wz, const float* Wc, const float* Wu,
                       const float* br, const float* bz, const float* bc, const float* bu,
                       float* gH) {
  __shared__ float sx[384];   // [shM | ehM | g_h]
  int b = blockIdx.x, j = threadIdx.x;
  sx[j]       = shM[(size_t)b * HH + j];
  sx[128 + j] = ehM[(size_t)b * HH + j];
  sx[256 + j] = gH[(size_t)b * HH + j];
  __syncthreads();
  float r = br[j], z = bz[j], c = bc[j], u0 = bu[j];
  for (int t = 0; t < 384; ++t) {
    float v = sx[t];
    r += v * Wr[(size_t)t * HH + j];
    z += v * Wz[(size_t)t * HH + j];
    if (t < 256) c  += v * Wc[(size_t)t * HH + j];
    else         u0 += v * Wu[(size_t)(t - 256) * HH + j];
  }
  r = sigf(r); z = sigf(z);
  float u = tanhf(c + r * u0);
  gH[(size_t)b * HH + j] = z * sx[256 + j] + (1.0f - z) * u;
}

// ---------------- host orchestration ----------------
extern "C" void kernel_launch(void* const* d_in, const int* in_sizes, int n_in,
                              void* d_out, int out_size, void* d_ws, size_t ws_size,
                              hipStream_t stream) {
  (void)in_sizes; (void)n_in; (void)out_size; (void)ws_size;
  // Inputs: sent, smask, word, wmask, matrix, then params flattened as a
  // jax pytree (dicts in sorted-key order, recursively):
  //   e_gru{Wc,Wr,Wu,Wz,bc,br,bu,bz}, edgeemb, g_gru{...}, gate1_W, gate1_b,
  //   gate2_W, gate2_b, s_gru{Wc,Wr,Wu,Wz,bc,br,bu,bz}
  const float* sent   = (const float*)d_in[0];
  const int*   smask  = (const int*)  d_in[1];
  const float* word   = (const float*)d_in[2];
  const int*   wmask  = (const int*)  d_in[3];
  const int*   matrix = (const int*)  d_in[4];
  const float* eWc = (const float*)d_in[5];
  const float* eWr = (const float*)d_in[6];
  const float* eWu = (const float*)d_in[7];
  const float* eWz = (const float*)d_in[8];
  const float* ebc = (const float*)d_in[9];
  const float* ebr = (const float*)d_in[10];
  const float* ebu = (const float*)d_in[11];
  const float* ebz = (const float*)d_in[12];
  const float* edgeemb = (const float*)d_in[13];
  const float* gWc = (const float*)d_in[14];
  const float* gWr = (const float*)d_in[15];
  const float* gWu = (const float*)d_in[16];
  const float* gWz = (const float*)d_in[17];
  const float* gbc = (const float*)d_in[18];
  const float* gbr = (const float*)d_in[19];
  const float* gbu = (const float*)d_in[20];
  const float* gbz = (const float*)d_in[21];
  const float* g1W = (const float*)d_in[22];
  const float* g1b = (const float*)d_in[23];
  const float* g2W = (const float*)d_in[24];
  const float* g2b = (const float*)d_in[25];
  const float* sWc = (const float*)d_in[26];
  const float* sWr = (const float*)d_in[27];
  const float* sWu = (const float*)d_in[28];
  const float* sWz = (const float*)d_in[29];
  const float* sbc = (const float*)d_in[30];
  const float* sbr = (const float*)d_in[31];
  const float* sbu = (const float*)d_in[32];
  const float* sbz = (const float*)d_in[33];

  // workspace bump allocator (256B aligned)
  char* base = (char*)d_ws;
  size_t off = 0;
  auto alloc = [&](size_t bytes) -> void* {
    off = (off + 255) & ~(size_t)255;
    void* p = base + off;
    off += bytes;
    return p;
  };
  // transposed bf16 weights
  __bf16* g1t  = (__bf16*)alloc((size_t)NG * 288 * 2);
  __bf16* g2t  = (__bf16*)alloc((size_t)NG * 288 * 2);
  __bf16* sWrT = (__bf16*)alloc((size_t)HH * RZS * 2);
  __bf16* sWzT = (__bf16*)alloc((size_t)HH * RZS * 2);
  __bf16* sWcT = (__bf16*)alloc((size_t)HH * XS * 2);
  __bf16* sWuT = (__bf16*)alloc((size_t)HH * HH * 2);
  __bf16* eWrT = (__bf16*)alloc((size_t)HH * RZE * 2);
  __bf16* eWzT = (__bf16*)alloc((size_t)HH * RZE * 2);
  __bf16* eWcT = (__bf16*)alloc((size_t)HH * XE * 2);
  __bf16* eWuT = (__bf16*)alloc((size_t)HH * HH * 2);
  float* T1 = (float*)alloc(4 * NG * 4);
  float* T2 = (float*)alloc(4 * NG * 4);
  // states (ping-pong)
  const int NS = BB * SS * HH, NE = BB * WW * HH;
  float*  sH[2]  = {(float*)alloc(NS * 4), (float*)alloc(NS * 4)};
  float*  eH[2]  = {(float*)alloc(NE * 4), (float*)alloc(NE * 4)};
  __bf16* sHb[2] = {(__bf16*)alloc(NS * 2), (__bf16*)alloc(NS * 2)};
  __bf16* eHb[2] = {(__bf16*)alloc(NE * 2), (__bf16*)alloc(NE * 2)};
  float*  gH = (float*)alloc(BB * HH * 4);
  unsigned* rowbits = (unsigned*)alloc(BB * SS * 4 * 4);
  unsigned* s2sbits = (unsigned*)alloc(BB * SS * 2 * 4);
  float* P1s = (float*)alloc((size_t)BB * SS * NG * 4);
  float* P1e = (float*)alloc((size_t)BB * WW * NG * 4);
  float* P2s = (float*)alloc((size_t)BB * SS * NG * 4);
  float* P2e = (float*)alloc((size_t)BB * WW * NG * 4);
  float* sns = (float*)alloc((size_t)BB * SS * HH * 4);
  float* sne = (float*)alloc((size_t)BB * SS * NG * 4);
  float* ens = (float*)alloc((size_t)BB * WW * NG * 4);
  __bf16* rzs = (__bf16*)alloc((size_t)BB * SS * RZS * 2);
  __bf16* rze = (__bf16*)alloc((size_t)BB * WW * RZE * 2);
  float* Rs  = (float*)alloc(NS * 4); float* Zs  = (float*)alloc(NS * 4);
  float* Cs  = (float*)alloc(NS * 4); float* U0s = (float*)alloc(NS * 4);
  float* Re  = (float*)alloc(NE * 4); float* Ze  = (float*)alloc(NE * 4);
  float* Ce  = (float*)alloc(NE * 4); float* U0e = (float*)alloc(NE * 4);
  float* shM = (float*)alloc(BB * HH * 4);
  float* ehM = (float*)alloc(BB * HH * 4);

  auto cdiv = [](int a, int b) { return (a + b - 1) / b; };
  auto gemm = [&](const __bf16* A, int lda, const __bf16* Wt, int ldw,
                  const float* bias, float* C, int ldc, int M, int N, int K, int act) {
    int tnGroups = cdiv(N / 16, 4);
    int blocks = (M / 64) * tnGroups;
    k_wmma_gemm<<<blocks, 128, 0, stream>>>(A, lda, Wt, ldw, bias, C, ldc, M, N, K, act);
  };

  // ---- one-time setup ----
  k_zero_f32<<<cdiv(NS, 256), 256, 0, stream>>>(sH[0], NS);
  k_zero_f32<<<cdiv(NE, 256), 256, 0, stream>>>(eH[0], NE);
  k_zero_f32<<<cdiv(BB * HH, 256), 256, 0, stream>>>(gH, BB * HH);
  k_zero_bf16<<<cdiv(NS, 256), 256, 0, stream>>>(sHb[0], NS);
  k_zero_bf16<<<cdiv(NE, 256), 256, 0, stream>>>(eHb[0], NE);
  k_transpose_bf16<<<cdiv(288 * NG, 256), 256, 0, stream>>>(g1W, g1t, 288, NG);
  k_transpose_bf16<<<cdiv(288 * NG, 256), 256, 0, stream>>>(g2W, g2t, 288, NG);
  k_transpose_bf16<<<cdiv(RZS * HH, 256), 256, 0, stream>>>(sWr, sWrT, RZS, HH);
  k_transpose_bf16<<<cdiv(RZS * HH, 256), 256, 0, stream>>>(sWz, sWzT, RZS, HH);
  k_transpose_bf16<<<cdiv(XS * HH, 256), 256, 0, stream>>>(sWc, sWcT, XS, HH);
  k_transpose_bf16<<<cdiv(HH * HH, 256), 256, 0, stream>>>(sWu, sWuT, HH, HH);
  k_transpose_bf16<<<cdiv(RZE * HH, 256), 256, 0, stream>>>(eWr, eWrT, RZE, HH);
  k_transpose_bf16<<<cdiv(RZE * HH, 256), 256, 0, stream>>>(eWz, eWzT, RZE, HH);
  k_transpose_bf16<<<cdiv(XE * HH, 256), 256, 0, stream>>>(eWc, eWcT, XE, HH);
  k_transpose_bf16<<<cdiv(HH * HH, 256), 256, 0, stream>>>(eWu, eWuT, HH, HH);
  k_label_table<<<cdiv(4 * NG, 128), 128, 0, stream>>>(edgeemb, g1W, g1b, T1);
  k_label_table<<<cdiv(4 * NG, 128), 128, 0, stream>>>(edgeemb, g2W, g2b, T2);
  k_rowbits<<<cdiv(BB * SS * 4, 256), 256, 0, stream>>>(matrix, rowbits);
  k_s2sbits<<<cdiv(BB * SS * 2, 256), 256, 0, stream>>>(rowbits, smask, s2sbits);

  // ---- layers ----
  int cur = 0;
  for (int layer = 0; layer < 2; ++layer) {
    int nxt = 1 - cur;
    k_neighbor_sum<<<BB * SS, HH, 0, stream>>>(s2sbits, sH[cur], sns);
    // gate projections (bias folded into T1/T2 -> none here)
    gemm(sHb[cur], HH, g1t, 288, nullptr, P1s, NG, BB * SS, NG, HH, 0);        // s_h @ G1[0:128]
    gemm(eHb[cur], HH, g1t + 128, 288, nullptr, P1e, NG, BB * WW, NG, HH, 0);  // e_h @ G1[128:256]
    gemm(eHb[cur], HH, g2t, 288, nullptr, P2e, NG, BB * WW, NG, HH, 0);        // e_h @ G2[0:128]
    gemm(sHb[cur], HH, g2t + 128, 288, nullptr, P2s, NG, BB * SS, NG, HH, 0);  // s_h @ G2[128:256]
    k_gather1<<<BB * SS, NG, 0, stream>>>(matrix, P1s, P1e, T1, eH[cur], edgeemb, sne);
    k_gather2<<<BB * WW, NG, 0, stream>>>(matrix, P2s, P2e, T2, sH[cur], edgeemb, ens);
    // build rz inputs (bf16)
    k_concat_rzs<<<cdiv(BB * SS * RZS, 256), 256, 0, stream>>>(sent, sns, sne, gH, sH[cur], rzs);
    k_concat_rze<<<cdiv(BB * WW * RZE, 256), 256, 0, stream>>>(word, ens, gH, eH[cur], rze);
    // sentence GRU matmuls
    gemm(rzs, RZS, sWrT, RZS, sbr, Rs, HH, BB * SS, HH, RZS, 1);
    gemm(rzs, RZS, sWzT, RZS, sbz, Zs, HH, BB * SS, HH, RZS, 1);
    gemm(rzs, RZS, sWcT, XS, sbc, Cs, HH, BB * SS, HH, XS, 0);   // x = first 544 cols of rz
    gemm(sHb[cur], HH, sWuT, HH, sbu, U0s, HH, BB * SS, HH, HH, 0);
    // edge GRU matmuls
    gemm(rze, RZE, eWrT, RZE, ebr, Re, HH, BB * WW, HH, RZE, 1);
    gemm(rze, RZE, eWzT, RZE, ebz, Ze, HH, BB * WW, HH, RZE, 1);
    gemm(rze, RZE, eWcT, XE, ebc, Ce, HH, BB * WW, HH, XE, 0);   // x = first 416 cols of rz
    gemm(eHb[cur], HH, eWuT, HH, ebu, U0e, HH, BB * WW, HH, HH, 0);
    // pointwise GRU + mask
    k_gru_combine<<<cdiv(NS, 256), 256, 0, stream>>>(Rs, Zs, Cs, U0s, sH[cur], smask, sH[nxt], sHb[nxt], NS);
    k_gru_combine<<<cdiv(NE, 256), 256, 0, stream>>>(Re, Ze, Ce, U0e, eH[cur], wmask, eH[nxt], eHb[nxt], NE);
    // means + global GRU
    k_mean<<<BB, HH, 0, stream>>>(sH[nxt], smask, shM, SS, 0.0f);
    k_mean<<<BB, HH, 0, stream>>>(eH[nxt], wmask, ehM, WW, 1.0f);
    k_ggru<<<BB, HH, 0, stream>>>(shM, ehM, gWr, gWz, gWc, gWu, gbr, gbz, gbc, gbu, gH);
    cur = nxt;
  }

  // ---- outputs: (s_h, e_h, g_h) concatenated flat ----
  float* out = (float*)d_out;
  hipMemcpyAsync(out,            sH[cur], (size_t)NS * 4, hipMemcpyDeviceToDevice, stream);
  hipMemcpyAsync(out + NS,       eH[cur], (size_t)NE * 4, hipMemcpyDeviceToDevice, stream);
  hipMemcpyAsync(out + NS + NE,  gH,      (size_t)BB * HH * 4, hipMemcpyDeviceToDevice, stream);
}
